// SwinEncoder_55310588838373
// MI455X (gfx1250) — matrix operand order repaired and checked by reading, not compiled, not run on hardware
//
#include <hip/hip_runtime.h>
#include <hip/hip_bf16.h>

// ---------------------------------------------------------------------------
// Swin encoder forward for MI455X (gfx1250), bf16 WMMA matmuls, fp32 residual.
//
// Input order assumed (setup_inputs dict insertion order, flattened):
//   0: x(64,3,224,224)
//   1..4: patch {w,b,ln_g,ln_b}
//   5..17 / 18..30: stage1 blocks (ln1_g,ln1_b,qkv_w,qkv_b,proj_w,proj_b,
//                                  rel_bias,ln2_g,ln2_b,fc1_w,fc1_b,fc2_w,fc2_b)
//   31,32,33: merge {ln_g,ln_b,red_w}
//   34..46 / 47..59: stage2 blocks
// ---------------------------------------------------------------------------

typedef __bf16 bf16;
typedef __attribute__((ext_vector_type(16))) __bf16 v16bf;
typedef __attribute__((ext_vector_type(8)))  __bf16 v8bf;
typedef __attribute__((ext_vector_type(8)))  float  v8f;

// ---- WMMA fragment helpers (CDNA5 16x16x32 bf16 layouts, wave32) -----------
// A (16x32, M x K): lane<16 -> row m=lane, K {0..7,16..23}; lane>=16 -> row
// m=lane-16, K {8..15,24..31}.  Two contiguous 8-elem chunks per lane.
__device__ __forceinline__ v16bf frag_a(const bf16* base, int ld, int row0, int k0) {
  int lane = threadIdx.x & 31;
  int m  = lane & 15;
  int kb = (lane < 16) ? 0 : 8;
  const bf16* p = base + (row0 + m) * ld + k0 + kb;
  v8bf lo = *(const v8bf*)(p);
  v8bf hi = *(const v8bf*)(p + 16);
  return __builtin_shufflevector(lo, hi, 0,1,2,3,4,5,6,7,8,9,10,11,12,13,14,15);
}
// B (32x16, K x N), sourced from row-major "W[n][k]": lane<16 -> col n=lane,
// K {0..7,16..23}; lane>=16 -> col n=lane-16, K {8..15,24..31}.  Two 8-elem reads.
__device__ __forceinline__ v16bf frag_b(const bf16* base, int ld, int col0, int k0) {
  int lane = threadIdx.x & 31;
  int n  = lane & 15;
  int kb = (lane < 16) ? 0 : 8;
  const bf16* p = base + (col0 + n) * ld + k0 + kb;
  v8bf lo = *(const v8bf*)(p);
  v8bf hi = *(const v8bf*)(p + 16);
  return __builtin_shufflevector(lo, hi, 0,1,2,3,4,5,6,7,8,9,10,11,12,13,14,15);
}
__device__ __forceinline__ v8f wmma_bf16(v16bf a, v16bf b, v8f c) {
  return __builtin_amdgcn_wmma_f32_16x16x32_bf16(false, a, false, b, (short)0, c,
                                                 false, false);
}

// ---- fp32 -> bf16 weight conversion ---------------------------------------
__global__ void cvt_bf16_kernel(const float* __restrict__ in, bf16* __restrict__ out, int n) {
  int i = blockIdx.x * 256 + threadIdx.x;
  if (i < n) out[i] = (bf16)in[i];
}

// ---- patch embedding conv 4x4 stride 4 (3->128) + bias, NHWC fp32 out -----
__global__ void patch_conv_kernel(const float* __restrict__ x, const float* __restrict__ w,
                                  const float* __restrict__ bias, float* __restrict__ out,
                                  int total) {
  int idx = blockIdx.x * 256 + threadIdx.x;
  if (idx >= total) return;
  int c   = idx & 127;
  int tok = idx >> 7;
  int ow  = tok % 56;
  int oh  = (tok / 56) % 56;
  int b   = tok / (56 * 56);
  float acc = bias[c];
  const float* wp = w + c * 48;              // (128,3,4,4)
  #pragma unroll
  for (int ci = 0; ci < 3; ++ci)
    #pragma unroll
    for (int kh = 0; kh < 4; ++kh) {
      const float* xp = x + ((size_t)(b * 3 + ci) * 224 + oh * 4 + kh) * 224 + ow * 4;
      const float* wr = wp + (ci * 4 + kh) * 4;
      #pragma unroll
      for (int kw = 0; kw < 4; ++kw) acc += xp[kw] * wr[kw];
    }
  out[idx] = acc;
}

// ---- LayerNorm: one wave per row, wave32 shuffle reduction ----------------
__global__ __launch_bounds__(128)
void ln_kernel(const float* __restrict__ in, const float* __restrict__ g,
               const float* __restrict__ b, bf16* __restrict__ outb,
               float* __restrict__ outf, int N, int C) {
  int row  = blockIdx.x * 4 + (threadIdx.x >> 5);
  int lane = threadIdx.x & 31;
  if (row >= N) return;
  const float* p = in + (size_t)row * C;
  float s = 0.f, s2 = 0.f;
  for (int c = lane; c < C; c += 32) { float v = p[c]; s += v; s2 += v * v; }
  #pragma unroll
  for (int off = 16; off > 0; off >>= 1) {
    s  += __shfl_xor(s,  off, 32);
    s2 += __shfl_xor(s2, off, 32);
  }
  float mean = s / (float)C;
  float var  = s2 / (float)C - mean * mean;
  float inv  = rsqrtf(var + 1e-5f);
  for (int c = lane; c < C; c += 32) {
    float v = (p[c] - mean) * inv * g[c] + b[c];
    if (outb) outb[(size_t)row * C + c] = (bf16)v;
    else      outf[(size_t)row * C + c] = v;
  }
}

// ---- patch merge: gather 2x2 neighborhood, concat(512), LN, bf16 out ------
__global__ __launch_bounds__(128)
void merge_ln_kernel(const float* __restrict__ R, const float* __restrict__ g,
                     const float* __restrict__ b, bf16* __restrict__ out, int B) {
  int t    = blockIdx.x * 4 + (threadIdx.x >> 5);
  int lane = threadIdx.x & 31;
  int total = B * 28 * 28;
  if (t >= total) return;
  int j  = t % 28, i = (t / 28) % 28, bb = t / (28 * 28);
  float vals[16];
  float s = 0.f, s2 = 0.f;
  #pragma unroll
  for (int u = 0; u < 16; ++u) {
    int c    = lane + u * 32;
    int quad = c >> 7, cc = c & 127;
    int dh = quad & 1, dw = quad >> 1;     // order x0(0,0) x1(1,0) x2(0,1) x3(1,1)
    float v = R[(((size_t)(bb * 56) + 2 * i + dh) * 56 + 2 * j + dw) * 128 + cc];
    vals[u] = v; s += v; s2 += v * v;
  }
  #pragma unroll
  for (int off = 16; off > 0; off >>= 1) {
    s  += __shfl_xor(s,  off, 32);
    s2 += __shfl_xor(s2, off, 32);
  }
  float mean = s / 512.f, var = s2 / 512.f - mean * mean;
  float inv  = rsqrtf(var + 1e-5f);
  #pragma unroll
  for (int u = 0; u < 16; ++u) {
    int c = lane + u * 32;
    out[(size_t)t * 512 + c] = (bf16)((vals[u] - mean) * inv * g[c] + b[c]);
  }
}

// ---- generic WMMA GEMM: Y[N][M] = X[N][K] * W[M][K]^T (+bias, epilogue) ---
// 128x128 tile, 256 threads = 8 waves; each wave computes 32x64 (2x4 frags).
// EPI 0: bias -> bf16 out;  1: bias + exact GELU -> bf16 out;
// EPI 2: bias + fp32 residual add -> fp32 out;  3: plain fp32 out (no bias).
template <int EPI>
__global__ __launch_bounds__(256)
void gemm_kernel(const bf16* __restrict__ X, const bf16* __restrict__ W,
                 const float* __restrict__ bias, const float* __restrict__ res,
                 bf16* __restrict__ outb, float* __restrict__ outf,
                 int N, int M, int K) {
  __shared__ bf16 Xs[128 * 32];
  __shared__ bf16 Ws[128 * 32];
  int n0 = blockIdx.y * 128, m0 = blockIdx.x * 128;
  int tid = threadIdx.x, lane = tid & 31, wave = tid >> 5;
  int wr = wave & 3, wc = wave >> 2;       // 4 row-bands x 2 col-bands
  int lrow = tid >> 1, lcol = (tid & 1) * 16;
  v8f acc[2][4] = {};
  for (int k0 = 0; k0 < K; k0 += 32) {
    const uint4* gx = (const uint4*)(X + (size_t)(n0 + lrow) * K + k0 + lcol);
    const uint4* gw = (const uint4*)(W + (size_t)(m0 + lrow) * K + k0 + lcol);
    uint4 x0 = gx[0], x1 = gx[1], w0 = gw[0], w1 = gw[1];
    if (k0 + 32 < K) {   // global_prefetch_b8 of next tiles
      __builtin_prefetch(X + (size_t)(n0 + lrow) * K + k0 + 32 + lcol, 0, 1);
      __builtin_prefetch(W + (size_t)(m0 + lrow) * K + k0 + 32 + lcol, 0, 1);
    }
    __syncthreads();
    *(uint4*)(Xs + lrow * 32 + lcol)     = x0;
    *(uint4*)(Xs + lrow * 32 + lcol + 8) = x1;
    *(uint4*)(Ws + lrow * 32 + lcol)     = w0;
    *(uint4*)(Ws + lrow * 32 + lcol + 8) = w1;
    __syncthreads();
    v16bf a0 = frag_a(Xs, 32, wr * 32,      0);
    v16bf a1 = frag_a(Xs, 32, wr * 32 + 16, 0);
    #pragma unroll
    for (int j = 0; j < 4; ++j) {
      v16bf bf = frag_b(Ws, 32, wc * 64 + j * 16, 0);
      acc[0][j] = wmma_bf16(a0, bf, acc[0][j]);
      acc[1][j] = wmma_bf16(a1, bf, acc[1][j]);
    }
  }
  int n = lane & 15, hi8 = (lane < 16) ? 0 : 8;
  #pragma unroll
  for (int i = 0; i < 2; ++i)
    #pragma unroll
    for (int j = 0; j < 4; ++j) {
      int col = m0 + wc * 64 + j * 16 + n;
      float bv = (EPI == 3) ? 0.f : bias[col];
      #pragma unroll
      for (int r = 0; r < 8; ++r) {
        int row = n0 + wr * 32 + i * 16 + hi8 + r;
        float v = acc[i][j][r] + bv;
        size_t o = (size_t)row * M + col;
        if (EPI == 0) {
          outb[o] = (bf16)v;
        } else if (EPI == 1) {
          v = 0.5f * v * (1.f + erff(v * 0.70710678118654752f));
          outb[o] = (bf16)v;
        } else if (EPI == 2) {
          outf[o] = res[o] + v;
        } else {
          outf[o] = v;
        }
      }
    }
}

// ---- shifted-window attention: one workgroup per (window, head), hd=32 ----
__device__ __forceinline__ int mregion(int p, int S, int shift) {
  if (shift == 0) return 0;
  if (p < S - 7) return 0;
  if (p < S - shift) return 1;
  return 2;
}

__global__ __launch_bounds__(128)
void attn_kernel(const bf16* __restrict__ qkv, const float* __restrict__ rel_bias,
                 bf16* __restrict__ AO, int Hs, int Wimg, int C, int heads, int shift) {
  __shared__ bf16 Qs[64 * 32];
  __shared__ bf16 Ks[64 * 32];
  __shared__ bf16 Vt[32 * 64];   // transposed: [d][token]
  __shared__ bf16 Ps[64 * 64];
  __shared__ float Sf[64 * 64];  // bias+mask table, then logits
  __shared__ int  tok[64];
  __shared__ int  rid[64];

  int nw = Wimg / 7, nh = Hs / 7;
  int win = blockIdx.x, h = blockIdx.y;
  int wj = win % nw, wi = (win / nw) % nh, b = win / (nw * nh);
  int tid = threadIdx.x, lane = tid & 31, wave = tid >> 5;
  int C3 = 3 * C;

  // stage Q/K (threads 0..63) and V^T (threads 64..127) into LDS, padded to 64
  {
    int r = tid & 63;
    int qi = r / 7, qj = r % 7;
    int rr = wi * 7 + qi, cc = wj * 7 + qj;               // rolled coords
    int oh = (rr + shift) % Hs, ow = (cc + shift) % Wimg; // undo roll(-shift)
    int t = (b * Hs + oh) * Wimg + ow;
    const bf16* base = qkv + (size_t)t * C3 + h * 32;
    if (tid < 64) {
      if (r < 49) {
        tok[r] = t;
        rid[r] = 3 * mregion(rr, Hs, shift) + mregion(cc, Wimg, shift);
        const uint4* qsrc = (const uint4*)(base);
        const uint4* ksrc = (const uint4*)(base + C);
        uint4* qdst = (uint4*)(Qs + r * 32);
        uint4* kdst = (uint4*)(Ks + r * 32);
        #pragma unroll
        for (int u = 0; u < 4; ++u) { qdst[u] = qsrc[u]; kdst[u] = ksrc[u]; }
      } else {
        tok[r] = 0; rid[r] = -1;
        uint4 z = {0, 0, 0, 0};
        uint4* qdst = (uint4*)(Qs + r * 32);
        uint4* kdst = (uint4*)(Ks + r * 32);
        #pragma unroll
        for (int u = 0; u < 4; ++u) { qdst[u] = z; kdst[u] = z; }
      }
    } else {
      const bf16* vsrc = base + 2 * C;
      if (r < 49) {
        #pragma unroll
        for (int d = 0; d < 32; ++d) Vt[d * 64 + r] = vsrc[d];
      } else {
        #pragma unroll
        for (int d = 0; d < 32; ++d) Vt[d * 64 + r] = (bf16)0.f;
      }
    }
  }
  __syncthreads();

  // pre-fill Sf with rel-pos bias + shift mask (+ -inf key padding)
  for (int e = tid; e < 64 * 64; e += 128) {
    int q = e >> 6, k = e & 63;
    float v;
    if (k >= 49) {
      v = -1e30f;
    } else if (q >= 49) {
      v = 0.f;
    } else {
      int qi = q / 7, qj = q % 7, ki = k / 7, kj = k % 7;
      int rel = (qi - ki + 6) * 13 + (qj - kj + 6);
      v = rel_bias[rel * heads + h];
      if (shift > 0 && rid[q] != rid[k]) v -= 100.f;
    }
    Sf[e] = v;
  }
  __syncthreads();

  // S = (Q * hd^-0.5) K^T + table ; wave handles 16 q-rows (branch-free epi)
  {
    v16bf a = frag_a(Qs, 32, wave * 16, 0);
    int n = lane & 15, hi8 = (lane < 16) ? 0 : 8;
    #pragma unroll
    for (int jc = 0; jc < 4; ++jc) {
      v16bf bf = frag_b(Ks, 32, jc * 16, 0);
      v8f acc = {};
      acc = wmma_bf16(a, bf, acc);
      int k = jc * 16 + n;
      #pragma unroll
      for (int r = 0; r < 8; ++r) {
        int q = wave * 16 + hi8 + r;
        Sf[q * 64 + k] += acc[r] * 0.17677669529663687f;  // 1/sqrt(32)
      }
    }
  }
  __syncthreads();

  // fp32 softmax, 2 threads per row (lane-pair shuffle combine), P -> bf16
  {
    int row = tid >> 1, half = tid & 1;
    const float* Sr = Sf + row * 64 + half * 32;
    float mx = -1e30f;
    #pragma unroll
    for (int k = 0; k < 32; ++k) mx = fmaxf(mx, Sr[k]);
    mx = fmaxf(mx, __shfl_xor(mx, 1, 32));
    float s = 0.f;
    #pragma unroll
    for (int k = 0; k < 32; ++k) s += __expf(Sr[k] - mx);
    s += __shfl_xor(s, 1, 32);
    float inv = 1.f / s;
    bf16* Pr = Ps + row * 64 + half * 32;
    #pragma unroll
    for (int k = 0; k < 32; ++k) Pr[k] = (bf16)(__expf(Sr[k] - mx) * inv);
  }
  __syncthreads();

  // O = P V  (K = 64 in two x32 steps), scatter back to token-major bf16
  v8f oacc[2] = {};
  #pragma unroll
  for (int ks = 0; ks < 64; ks += 32) {
    v16bf a = frag_a(Ps, 64, wave * 16, ks);
    oacc[0] = wmma_bf16(a, frag_b(Vt, 64, 0,  ks), oacc[0]);
    oacc[1] = wmma_bf16(a, frag_b(Vt, 64, 16, ks), oacc[1]);
  }
  int n = lane & 15, hi8 = (lane < 16) ? 0 : 8;
  #pragma unroll
  for (int j = 0; j < 2; ++j)
    #pragma unroll
    for (int r = 0; r < 8; ++r) {
      int q = wave * 16 + hi8 + r;
      if (q < 49)
        AO[(size_t)tok[q] * C + h * 32 + j * 16 + n] = (bf16)oacc[j][r];
    }
}

// ---------------------------------------------------------------------------
extern "C" void kernel_launch(void* const* d_in, const int* in_sizes, int n_in,
                              void* d_out, int out_size, void* d_ws, size_t ws_size,
                              hipStream_t stream) {
  (void)in_sizes; (void)n_in; (void)out_size; (void)ws_size;
  const float* x   = (const float*)d_in[0];
  const float* pw  = (const float*)d_in[1];
  const float* pb  = (const float*)d_in[2];
  const float* plg = (const float*)d_in[3];
  const float* plb = (const float*)d_in[4];

  const int N1 = 64 * 56 * 56;  // 200704 tokens, C=128
  const int N2 = 64 * 28 * 28;  //  50176 tokens, C=256

  // workspace carve-out (aligned, reused across phases; ~420 MB total)
  char* ws = (char*)d_ws;
  size_t off = 0;
  auto carve = [&](size_t bytes) {
    void* p = (void*)(ws + off);
    off = (off + bytes + 255) & ~(size_t)255;
    return p;
  };
  float* R   = (float*)carve((size_t)N1 * 128 * 4);  // fp32 residual stream
  bf16*  XLN = (bf16*) carve((size_t)N1 * 128 * 2);  // LN output (also merge out)
  bf16*  BIG = (bf16*) carve((size_t)N1 * 512 * 2);  // qkv out / fc1 out / conv tmp
  bf16*  AO  = (bf16*) carve((size_t)N1 * 128 * 2);  // attention output
  bf16*  WB  = (bf16*) carve((size_t)8 * 1024 * 1024); // bf16 weight pool

  size_t wboff = 0;
  auto cvt = [&](const float* src, int n) {
    bf16* dst = WB + wboff;
    wboff += (size_t)((n + 127) & ~127);
    cvt_bf16_kernel<<<(n + 255) / 256, 256, 0, stream>>>(src, dst, n);
    return dst;
  };

  // one full swin block: x += attn(LN1 x); x += fc2(gelu(fc1(LN2 x)))
  auto run_block = [&](int base_idx, int C, int heads, int Himg, int shift,
                       int Ntok, float* fc2_out) {
    const float* ln1g  = (const float*)d_in[base_idx + 0];
    const float* ln1b  = (const float*)d_in[base_idx + 1];
    const float* qkvw  = (const float*)d_in[base_idx + 2];
    const float* qkvb  = (const float*)d_in[base_idx + 3];
    const float* projw = (const float*)d_in[base_idx + 4];
    const float* projb = (const float*)d_in[base_idx + 5];
    const float* relb  = (const float*)d_in[base_idx + 6];
    const float* ln2g  = (const float*)d_in[base_idx + 7];
    const float* ln2b  = (const float*)d_in[base_idx + 8];
    const float* fc1w  = (const float*)d_in[base_idx + 9];
    const float* fc1b  = (const float*)d_in[base_idx + 10];
    const float* fc2w  = (const float*)d_in[base_idx + 11];
    const float* fc2b  = (const float*)d_in[base_idx + 12];

    bf16* wqkv  = cvt(qkvw, 3 * C * C);
    bf16* wproj = cvt(projw, C * C);
    bf16* wfc1  = cvt(fc1w, 4 * C * C);
    bf16* wfc2  = cvt(fc2w, 4 * C * C);

    ln_kernel<<<Ntok / 4, 128, 0, stream>>>(R, ln1g, ln1b, XLN, nullptr, Ntok, C);
    gemm_kernel<0><<<dim3(3 * C / 128, Ntok / 128), 256, 0, stream>>>(
        XLN, wqkv, qkvb, nullptr, BIG, nullptr, Ntok, 3 * C, C);
    int nwin = 64 * (Himg / 7) * (Himg / 7);
    attn_kernel<<<dim3(nwin, heads), 128, 0, stream>>>(
        BIG, relb, AO, Himg, Himg, C, heads, shift);
    gemm_kernel<2><<<dim3(C / 128, Ntok / 128), 256, 0, stream>>>(
        AO, wproj, projb, R, nullptr, R, Ntok, C, C);
    ln_kernel<<<Ntok / 4, 128, 0, stream>>>(R, ln2g, ln2b, XLN, nullptr, Ntok, C);
    gemm_kernel<1><<<dim3(4 * C / 128, Ntok / 128), 256, 0, stream>>>(
        XLN, wfc1, fc1b, nullptr, BIG, nullptr, Ntok, 4 * C, C);
    gemm_kernel<2><<<dim3(C / 128, Ntok / 128), 256, 0, stream>>>(
        BIG, wfc2, fc2b, R, nullptr, fc2_out, Ntok, C, 4 * C);
  };

  // 1) patch embed: conv -> fp32 tmp (in BIG), LN -> R (fp32)
  float* TMP = (float*)BIG;
  patch_conv_kernel<<<(N1 * 128 + 255) / 256, 256, 0, stream>>>(
      x, pw, pb, TMP, N1 * 128);
  ln_kernel<<<N1 / 4, 128, 0, stream>>>(TMP, plg, plb, nullptr, R, N1, 128);

  // 2) stage 1 (C=128, heads=4, 56x56; shifts 0 and 3)
  run_block(5,  128, 4, 56, 0, N1, R);
  run_block(18, 128, 4, 56, 3, N1, R);

  // 3) patch merge: gather+LN -> XLN (50176x512 bf16), GEMM red_w -> R (fp32)
  merge_ln_kernel<<<N2 / 4, 128, 0, stream>>>(
      R, (const float*)d_in[31], (const float*)d_in[32], XLN, 64);
  bf16* wred = cvt((const float*)d_in[33], 256 * 512);
  gemm_kernel<3><<<dim3(256 / 128, N2 / 128), 256, 0, stream>>>(
      XLN, wred, nullptr, nullptr, nullptr, R, N2, 256, 512);

  // 4) stage 2 (C=256, heads=8, 28x28); last fc2 writes straight to d_out
  run_block(34, 256, 8, 28, 0, N2, R);
  run_block(47, 256, 8, 28, 3, N2, (float*)d_out);
}
